// MultiheadAttentionMechanism_68942815035879
// MI455X (gfx1250) — matrix operand order repaired and checked by reading, not compiled, not run on hardware
//
#include <hip/hip_runtime.h>
#include <hip/hip_bf16.h>
#include <math.h>

typedef __attribute__((ext_vector_type(16))) _Float16 v16h;
typedef __attribute__((ext_vector_type(8)))  float    v8f;

#define NEG_INF_F (-3.402823466e38f)

// ---------------------------------------------------------------------------
// 8-element vector load / convert / LDS-store helpers
// ---------------------------------------------------------------------------
template <typename T> struct alignas(16) Pack8 { T v[8]; };

__device__ inline Pack8<float> load8(const float* __restrict__ p) {
    Pack8<float> r;
    const float4 a = *reinterpret_cast<const float4*>(p);
    const float4 b = *reinterpret_cast<const float4*>(p + 4);
    r.v[0] = a.x; r.v[1] = a.y; r.v[2] = a.z; r.v[3] = a.w;
    r.v[4] = b.x; r.v[5] = b.y; r.v[6] = b.z; r.v[7] = b.w;
    return r;
}
__device__ inline Pack8<_Float16> load8(const _Float16* __restrict__ p) {
    Pack8<_Float16> r;
    *reinterpret_cast<uint4*>(r.v) = *reinterpret_cast<const uint4*>(p);
    return r;
}
__device__ inline void store8_lds(_Float16* d, const Pack8<float>& x) {
    union { _Float16 h[8]; uint2 u[2]; } t;
    #pragma unroll
    for (int i = 0; i < 8; ++i) t.h[i] = (_Float16)x.v[i];
    reinterpret_cast<uint2*>(d)[0] = t.u[0];
    reinterpret_cast<uint2*>(d)[1] = t.u[1];
}
__device__ inline void store8_lds(_Float16* d, const Pack8<_Float16>& x) {
    const uint2* s = reinterpret_cast<const uint2*>(x.v);
    reinterpret_cast<uint2*>(d)[0] = s[0];
    reinterpret_cast<uint2*>(d)[1] = s[1];
}

// ---------------------------------------------------------------------------
// Generic WMMA GEMM:  C[z] = alpha * A[z] * B[z] (+ bias)
//   Block tile 64(M) x 64(N), staged K-chunk 64 (2 WMMA k-steps of 32),
//   256 threads = 8 waves. Wave w: wm = w&3 -> 16-row M subtile,
//   wn = w>>2 -> 32-col N slab; 4 v_wmma_f32_16x16x32_f16 per barrier pair.
//   Staging: register double buffer, 16 contiguous elems/thread,
//   b128 global loads, f32->f16 convert in regs, b64 LDS stores.
//   All dims assumed multiples of 64 (true for every call here).
// ---------------------------------------------------------------------------
template <typename AT, typename BT, typename OT, bool HASBIAS>
__global__ __launch_bounds__(256)
void gemm_wmma(const AT* __restrict__ A, long lda, long sAo, long sAi,
               const BT* __restrict__ Bm, long ldb, long sBo, long sBi,
               const float* __restrict__ bias,
               OT* __restrict__ C, long ldc, long sCo, long sCi,
               int K, int zdiv, float alpha)
{
    constexpr int LDA_S = 68;   // 64 + 4 pad halves -> conflict-free frag reads
    constexpr int LDB_S = 68;
    __shared__ _Float16 ldsA[64 * LDA_S];
    __shared__ _Float16 ldsB[64 * LDB_S];

    const int z  = blockIdx.z;
    const int zo = z / zdiv, zi = z % zdiv;
    A  += (long)zo * sAo + (long)zi * sAi;
    Bm += (long)zo * sBo + (long)zi * sBi;
    C  += (long)zo * sCo + (long)zi * sCi;

    const int m0 = blockIdx.x * 64;
    const int n0 = blockIdx.y * 64;

    const int tid     = threadIdx.x;
    const int lane    = tid & 31;
    const int wave    = tid >> 5;
    const int wm      = wave & 3;     // M subtile
    const int wn      = wave >> 2;    // N slab (32 cols)
    const int halfsel = lane >> 4;
    const int mrow    = lane & 15;

    // staging: 4 threads per 64-elem row, 16 contiguous elems each
    const int ar  = tid >> 2;
    const int ac0 = (tid & 3) << 4;
    const AT* Ap = A  + (long)(m0 + ar) * lda + ac0;
    const BT* Bp = Bm + (long)ar * ldb + n0 + ac0;
    _Float16* la = &ldsA[ar * LDA_S + ac0];
    _Float16* lb = &ldsB[ar * LDB_S + ac0];

    Pack8<AT> ra0 = load8(Ap), ra1 = load8(Ap + 8);
    Pack8<BT> rb0 = load8(Bp), rb1 = load8(Bp + 8);

    v8f acc0 = {};
    v8f acc1 = {};

    for (int k0 = 0; k0 < K; k0 += 64) {
        store8_lds(la,     ra0);
        store8_lds(la + 8, ra1);
        store8_lds(lb,     rb0);
        store8_lds(lb + 8, rb1);
        __syncthreads();

        // issue next chunk's global loads before consuming this one
        if (k0 + 64 < K) {
            const AT* An = Ap + (k0 + 64);
            const BT* Bn = Bp + (long)(k0 + 64) * ldb;
            ra0 = load8(An); ra1 = load8(An + 8);
            rb0 = load8(Bn); rb1 = load8(Bn + 8);
        }

        #pragma unroll
        for (int ks = 0; ks < 64; ks += 32) {
            // A fragment: 16x32 f16, lane = M row, ISA K layout
            union Frag { v16h v; unsigned u[8]; };
            Frag a;
            #pragma unroll
            for (int r = 0; r < 8; ++r) {
                int kk = ks + ((r >> 2) << 4) + (halfsel << 3) + ((r & 3) << 1);
                a.u[r] = *reinterpret_cast<const unsigned*>(
                             &ldsA[(wm * 16 + mrow) * LDA_S + kk]);
            }
            // B fragments: 32x16 f16, lane = N col, ISA K layout
            Frag b0, b1;
            #pragma unroll
            for (int s = 0; s < 16; ++s) {
                int kk = ks + ((s >> 3) << 4) + (halfsel << 3) + (s & 7);
                b0.v[s] = ldsB[kk * LDB_S + wn * 32 + mrow];
                b1.v[s] = ldsB[kk * LDB_S + wn * 32 + 16 + mrow];
            }
            acc0 = __builtin_amdgcn_wmma_f32_16x16x32_f16(
                       false, a.v, false, b0.v, (short)0, acc0, false, false);
            acc1 = __builtin_amdgcn_wmma_f32_16x16x32_f16(
                       false, a.v, false, b1.v, (short)0, acc1, false, false);
        }
        __syncthreads();
    }

    // store: D VGPR r -> (M = r + 8*halfsel, N = lane&15)
    const int gm = m0 + wm * 16 + (halfsel << 3);
    const int gn = n0 + wn * 32 + (lane & 15);
    #pragma unroll
    for (int r = 0; r < 8; ++r) {
        float v0 = alpha * acc0[r];
        float v1 = alpha * acc1[r];
        if (HASBIAS) { v0 += bias[gn]; v1 += bias[gn + 16]; }
        C[(long)(gm + r) * ldc + gn]      = (OT)v0;
        C[(long)(gm + r) * ldc + gn + 16] = (OT)v1;
    }
}

// ---------------------------------------------------------------------------
// Per-head transpose: khT[b,h,d,j] = kh[b,j,h*DK+d]  (f16, LDS-tiled)
// grid: (L/64, B*H), 256 threads. Tile 64(j) x 64(d).
// ---------------------------------------------------------------------------
__global__ __launch_bounds__(256)
void transpose_head(const _Float16* __restrict__ kh, _Float16* __restrict__ khT,
                    int H, int L, int DK, int AD)
{
    __shared__ _Float16 tile[64 * 68];
    const int bh = blockIdx.y;           // b*H + h
    const int b  = bh / H, h = bh % H;
    const int j0 = blockIdx.x * 64;
    const int t  = threadIdx.x;

    {   // load 64 j-rows x 64 d (contiguous): 4 threads/row, 16 halves each
        const int jj = t >> 2, seg = t & 3;
        const _Float16* src =
            kh + ((long)(b * L + j0 + jj) * AD + h * DK + seg * 16);
        Pack8<_Float16> p0 = load8(src);
        Pack8<_Float16> p1 = load8(src + 8);
        store8_lds(&tile[jj * 68 + seg * 16], p0);
        store8_lds(&tile[jj * 68 + seg * 16 + 8], p1);
    }
    __syncthreads();
    {   // store 64 d-rows x 64 j (contiguous out): 4 threads/row, 16 each
        const int d = t >> 2, seg = t & 3;
        union { _Float16 h16[16]; uint4 u[2]; } o;
        #pragma unroll
        for (int i = 0; i < 16; ++i)
            o.h16[i] = tile[(seg * 16 + i) * 68 + d];
        _Float16* dst = khT + ((long)bh * DK + d) * L + j0 + seg * 16;
        reinterpret_cast<uint4*>(dst)[0] = o.u[0];
        reinterpret_cast<uint4*>(dst)[1] = o.u[1];
    }
}

// ---------------------------------------------------------------------------
// Masked softmax over klen (L=1024), in place on aw (f32).
// One 256-thread block per (b,h,i) row; one float4 per thread.
// ---------------------------------------------------------------------------
__global__ __launch_bounds__(256)
void softmax_mask_kernel(float* __restrict__ aw, const int* __restrict__ mask,
                         int H, int L)
{
    const int x = blockIdx.x;
    const int i = x % L;
    const int h = (x / L) % H;
    const int b = x / (L * H);

    float* row      = aw   + ((long)(b * H + h) * L + i) * L;
    const int* mrow = mask + ((long)b * L + i) * L;

    __shared__ float sred[256];
    const int tid = threadIdx.x;

    float4 ev = reinterpret_cast<const float4*>(row)[tid];
    int4   mv = reinterpret_cast<const int4*>(mrow)[tid];
    float vals[4] = { mv.x ? ev.x : NEG_INF_F, mv.y ? ev.y : NEG_INF_F,
                      mv.z ? ev.z : NEG_INF_F, mv.w ? ev.w : NEG_INF_F };
    float mx = fmaxf(fmaxf(vals[0], vals[1]), fmaxf(vals[2], vals[3]));

    sred[tid] = mx;
    __syncthreads();
    for (int s = 128; s > 0; s >>= 1) {
        if (tid < s) sred[tid] = fmaxf(sred[tid], sred[tid + s]);
        __syncthreads();
    }
    mx = sred[0];
    __syncthreads();

    float sum = 0.0f;
    #pragma unroll
    for (int t = 0; t < 4; ++t) { vals[t] = expf(vals[t] - mx); sum += vals[t]; }
    sred[tid] = sum;
    __syncthreads();
    for (int s = 128; s > 0; s >>= 1) {
        if (tid < s) sred[tid] += sred[tid + s];
        __syncthreads();
    }
    const float inv = 1.0f / sred[0];
    float4 o = { vals[0] * inv, vals[1] * inv, vals[2] * inv, vals[3] * inv };
    reinterpret_cast<float4*>(row)[tid] = o;
}

// ---------------------------------------------------------------------------
// Host-side orchestration.
// Inputs: 0 key, 1 value, 2 query, 3 mask, 4 Wk, 5 bk, 6 Wv, 7 bv,
//         8 Wq, 9 bq, 10 Wo, 11 bo
// Outputs (concatenated f32): cv [B, L, 1024] (2M), aw [B, H, L, L] (32M)
// ---------------------------------------------------------------------------
extern "C" void kernel_launch(void* const* d_in, const int* in_sizes, int n_in,
                              void* d_out, int out_size, void* d_ws, size_t ws_size,
                              hipStream_t stream)
{
    (void)in_sizes; (void)n_in; (void)out_size; (void)ws_size;

    constexpr int  Bc = 2, Hc = 16, Lc = 1024, AD = 1024, DK = 64;
    constexpr long NTOK = (long)Bc * Lc;           // 2048 rows
    constexpr long ACT  = (long)Bc * Lc * AD;      // 2M elements

    const float* key   = (const float*)d_in[0];
    const float* value = (const float*)d_in[1];
    const float* query = (const float*)d_in[2];
    const int*   mask  = (const int*)  d_in[3];
    const float* Wk    = (const float*)d_in[4];
    const float* bk    = (const float*)d_in[5];
    const float* Wv    = (const float*)d_in[6];
    const float* bv    = (const float*)d_in[7];
    const float* Wq    = (const float*)d_in[8];
    const float* bq    = (const float*)d_in[9];
    const float* Wo    = (const float*)d_in[10];
    const float* bo    = (const float*)d_in[11];

    float* out    = (float*)d_out;
    float* cv_out = out;                     // [2048 x 1024]
    float* aw     = out + ACT;               // [B, H, L, L]

    _Float16* qh  = (_Float16*)d_ws;         // f16 buffers, 4 MB each
    _Float16* kh  = qh  + ACT;
    _Float16* vh  = kh  + ACT;
    _Float16* cvh = vh  + ACT;
    _Float16* khT = cvh + ACT;               // [B, H, DK, L]

    dim3 blk(256);
    dim3 gproj(NTOK / 64, AD / 64, 1);

    // --- Q/K/V projections (f32 in, f16 out): X @ W + b ---
    gemm_wmma<float, float, _Float16, true><<<gproj, blk, 0, stream>>>(
        key,   AD, 0, 0, Wk, AD, 0, 0, bk, kh, AD, 0, 0, AD, 1, 1.0f);
    gemm_wmma<float, float, _Float16, true><<<gproj, blk, 0, stream>>>(
        value, AD, 0, 0, Wv, AD, 0, 0, bv, vh, AD, 0, 0, AD, 1, 1.0f);
    gemm_wmma<float, float, _Float16, true><<<gproj, blk, 0, stream>>>(
        query, AD, 0, 0, Wq, AD, 0, 0, bq, qh, AD, 0, 0, AD, 1, 1.0f);

    // --- Transpose K per head: khT[b,h,d,j] ---
    dim3 gtr(Lc / 64, Bc * Hc, 1);
    transpose_head<<<gtr, blk, 0, stream>>>(kh, khT, Hc, Lc, DK, AD);

    // --- Scores: per (b,h) S = (Q @ K^T) / 8, raw f32 into aw region ---
    // K = 64 -> single staged chunk, one barrier pair per block.
    dim3 gsc(Lc / 64, Lc / 64, Bc * Hc);
    gemm_wmma<_Float16, _Float16, float, false><<<gsc, blk, 0, stream>>>(
        qh,  AD, (long)Lc * AD, DK,
        khT, Lc, (long)Hc * DK * Lc, (long)DK * Lc, nullptr,
        aw,  Lc, (long)Hc * Lc * Lc, (long)Lc * Lc,
        DK, Hc, 1.0f / 8.0f);

    // --- Masked softmax over klen, in place (aw becomes final output) ---
    softmax_mask_kernel<<<Bc * Hc * Lc, blk, 0, stream>>>(aw, mask, Hc, Lc);

    // --- Context: per (b,h) cv = P @ V  (P f32 -> f16 during staging) ---
    dim3 gctx(Lc / 64, DK / 64, Bc * Hc);
    gemm_wmma<float, _Float16, _Float16, false><<<gctx, blk, 0, stream>>>(
        aw, Lc, (long)Hc * Lc * Lc, (long)Lc * Lc,
        vh, AD, (long)Lc * AD, DK, nullptr,
        cvh, AD, (long)Lc * AD, DK,
        Lc, Hc, 1.0f);

    // --- Output projection: cv @ Wo + bo (f32 out) ---
    gemm_wmma<_Float16, float, float, true><<<gproj, blk, 0, stream>>>(
        cvh, AD, 0, 0, Wo, AD, 0, 0, bo, cv_out, AD, 0, 0, AD, 1, 1.0f);
}
